// EmformerLayer_25666724561393
// MI455X (gfx1250) — matrix operand order repaired and checked by reading, not compiled, not run on hardware
//
#include <hip/hip_runtime.h>
#include <hip/hip_bf16.h>

// ---------------------------------------------------------------------------
// Emformer layer for MI455X (gfx1250), wave32 + v_wmma_f32_16x16x32_f16.
// Round 4: preload all 4 B fragments per k-step in the GEMM so loads clause
//          and waits stagger instead of s_wait_loadcnt 0x0 per WMMA.
//
// Shapes: U=1024 B=8 D=512 H=8 HD=64 R=64 S=16 M=15 Q=1104 KV=1103 PE=2047
// ---------------------------------------------------------------------------

typedef __attribute__((ext_vector_type(16))) _Float16 v16h;
typedef __attribute__((ext_vector_type(8)))  float    v8f;
typedef __attribute__((ext_vector_type(4)))  int      v4i;

#define QLEN   1104      // R + U + S
#define KVLEN  1103      // M + R + U
#define KVPAD  1120      // padded to 35 K32 tiles for P@V
#define SLD    1120      // LDS score row stride (floats)
#define PEPAD  2048
#define NEG_INF_VAL (-100000000.0f)

#if __has_builtin(__builtin_amdgcn_global_load_async_to_lds_b128) && \
    __has_builtin(__builtin_amdgcn_s_wait_asynccnt)
#define USE_ASYNC_LDS 1
#else
#define USE_ASYNC_LDS 0
#endif

__device__ __forceinline__ v8f wmma_f16f32(v16h a, v16h b, v8f c) {
  // D = A(16x32 f16) * B(32x16 f16) + C(16x16 f32)
  return __builtin_amdgcn_wmma_f32_16x16x32_f16(false, a, false, b,
                                                (short)0, c, false, false);
}

#if USE_ASYNC_LDS
// Async 16-byte global -> LDS copy (tracked by ASYNCcnt).
__device__ __forceinline__ void async_copy16(const _Float16* gsrc, _Float16* ldst) {
  v4i* gg = (v4i*)const_cast<_Float16*>(gsrc);   // bitcast in generic AS
  v4i* ll = (v4i*)ldst;
  __builtin_amdgcn_global_load_async_to_lds_b128(
      (__attribute__((address_space(1))) v4i*)gg,  // addrspacecast -> global
      (__attribute__((address_space(3))) v4i*)ll,  // addrspacecast -> LDS
      0, 0);
}
#endif

// A fragment 16x32 (f16), src row-major [M][ld].
// lane L: row = m0 + L%16 ; halves 0..7 = K k0+(L/16)*8+i ; halves 8..15 = +16.
__device__ __forceinline__ v16h load_A(const _Float16* src, int ld, int m0,
                                       int k0, int lane) {
  const _Float16* p = src + (size_t)(m0 + (lane & 15)) * ld + k0 + ((lane >> 4) << 3);
  v16h a;
#pragma unroll
  for (int i = 0; i < 8; ++i) { a[i] = p[i]; a[i + 8] = p[i + 16]; }
  return a;
}

// B fragment 32x16 with B[k][n] = src[n][k]  (i.e. multiply by src^T),
// src row-major [N][ld]. lane L: col = n0 + L%16 ; halves 0..7 = K k0+(L/16)*8+i ; halves 8..15 = +16.
__device__ __forceinline__ v16h load_B_T(const _Float16* src, int ld, int n0,
                                         int k0, int lane) {
  const _Float16* p = src + (size_t)(n0 + (lane & 15)) * ld + k0 + ((lane >> 4) << 3);
  v16h b;
#pragma unroll
  for (int i = 0; i < 8; ++i) { b[i] = p[i]; b[i + 8] = p[i + 16]; }
  return b;
}

// ---------------------------------------------------------------------------
// Generic GEMM: out[M][N] = X[M][K](f16) @ W[N][K](f16)^T + bias[N], f32 out.
// One wave per 16x64 tile: one A-fragment load feeds 4 WMMAs per K-step.
// All 5 fragment loads are issued before the 4 WMMAs so VMEM overlaps XDL.
// ---------------------------------------------------------------------------
__global__ __launch_bounds__(32) void gemm_xwT(const _Float16* __restrict__ X,
                                               const _Float16* __restrict__ W,
                                               const float* __restrict__ bias,
                                               float* __restrict__ out,
                                               int N, int K) {
  int lane = threadIdx.x & 31;
  int m0 = blockIdx.x * 16, n0 = blockIdx.y * 64;
  v8f acc[4] = {{}, {}, {}, {}};
  for (int k0 = 0; k0 < K; k0 += 32) {
    v16h a = load_A(X, K, m0, k0, lane);
    v16h bfr[4];
#pragma unroll
    for (int s = 0; s < 4; ++s) bfr[s] = load_B_T(W, K, n0 + s * 16, k0, lane);
#pragma unroll
    for (int s = 0; s < 4; ++s) acc[s] = wmma_f16f32(a, bfr[s], acc[s]);
  }
  int rb = m0 + ((lane >> 4) << 3);
#pragma unroll
  for (int s = 0; s < 4; ++s) {
    int col = n0 + s * 16 + (lane & 15);
    float bv = bias ? bias[col] : 0.0f;
#pragma unroll
    for (int j = 0; j < 8; ++j) out[(size_t)(rb + j) * N + col] = acc[s][j] + bv;
  }
}

// ---------------------------------------------------------------------------
// Fused attention: per (b,h, 16-row q tile):
//   scores (AC via WMMA, K tiles async-staged to LDS) +
//   rel-pos BD (WMMA on pos window + diagonal scatter) -> mask -> softmax ->
//   probs(f16) @ V^T (WMMA) -> attn[q][b][d]
// Dynamic LDS: sS[16][SLD] f32  +  sP[16][SLD] f16
// Static  LDS: stage[4 waves][2 bufs][1024 halves] for async double-buffering
// ---------------------------------------------------------------------------
__global__ __launch_bounds__(128) void attention_kernel(
    const _Float16* __restrict__ QU, const _Float16* __restrict__ QV,
    const _Float16* __restrict__ Kh, const _Float16* __restrict__ VT,
    const _Float16* __restrict__ Ph, const int* __restrict__ lengths,
    float* __restrict__ attn_out) {
  extern __shared__ char smem[];
  float*    sS = (float*)smem;                        // 16*SLD f32
  _Float16* sP = (_Float16*)(smem + 16 * SLD * 4);    // 16*SLD f16

  int qt = blockIdx.x;                 // 0..68
  int bh = blockIdx.y;                 // b*8 + h
  int b = bh >> 3, h = bh & 7;
  int tid = threadIdx.x, wave = tid >> 5, lane = tid & 31;
  int q0 = qt * 16;

  const _Float16* qu = QU + (size_t)bh * QLEN * 64;
  const _Float16* qv = QV + (size_t)bh * 1024 * 64;
  const _Float16* kh = Kh + (size_t)bh * QLEN * 64;    // 1104 rows (row 1103 = 0)
  const _Float16* vt = VT + (size_t)bh * 64 * KVPAD;   // [64][1120], pad cols = 0
  const _Float16* ph = Ph + (size_t)h * PEPAD * 64;    // 2048 rows (row 2047 = 0)

#if USE_ASYNC_LDS
  __shared__ __align__(16) _Float16 stage[4][2][1024];  // 2KB tile per buffer
  // Stage one 16x64-half (2KB, contiguous) operand tile into LDS buffer `buf`.
  auto stage_tile = [&](const _Float16* src, int buf) {
#pragma unroll
    for (int i = 0; i < 4; ++i) {
      int off = (i * 32 + lane) * 8;   // halves; 16B per lane per issue
      async_copy16(src + off, &stage[wave][buf][off]);
    }
  };
#endif

  // ---- Phase 1: AC scores: sS[0:16][n0:n0+16] = qu_tile . k_tile^T --------
  v16h a0 = load_A(qu, 64, q0, 0, lane);
  v16h a1 = load_A(qu, 64, q0, 32, lane);
#if USE_ASYNC_LDS
  {
    int nt = 0;
    stage_tile(kh + (size_t)wave * 1024, 0);
    for (int t = wave; t < 69; t += 4) {
      if (t + 4 < 69) {
        stage_tile(kh + (size_t)(t + 4) * 1024, nt ^ 1);
        __builtin_amdgcn_s_wait_asynccnt(4);   // tile t complete (in-order)
      } else {
        __builtin_amdgcn_s_wait_asynccnt(0);
      }
      const _Float16* bsrc = &stage[wave][nt][0];
      v8f acc = {};
      acc = wmma_f16f32(a0, load_B_T(bsrc, 64, 0, 0, lane), acc);
      acc = wmma_f16f32(a1, load_B_T(bsrc, 64, 0, 32, lane), acc);
      int col = t * 16 + (lane & 15);
      int rb = (lane >> 4) << 3;
#pragma unroll
      for (int j = 0; j < 8; ++j) sS[(rb + j) * SLD + col] = acc[j];
      nt ^= 1;
    }
  }
#else
  for (int t = wave; t < 69; t += 4) {
    int n0 = t * 16;
    v8f acc = {};
    acc = wmma_f16f32(a0, load_B_T(kh, 64, n0, 0, lane), acc);
    acc = wmma_f16f32(a1, load_B_T(kh, 64, n0, 32, lane), acc);
    int col = n0 + (lane & 15);
    int rb = (lane >> 4) << 3;
#pragma unroll
    for (int j = 0; j < 8; ++j) sS[(rb + j) * SLD + col] = acc[j];
  }
#endif
  __syncthreads();

  // ---- Phase 2: rel-pos BD (utterance q-rows only, tiles 4..67) -----------
  // BD[i][j] = qv[i] . pos[U-1-i+j].  For row-tile i0 all needed pos rows
  // sit in window [w0, w0+1039), w0 = 1008-i0.  Compute qv_tile @ pos_win^T
  // with WMMA and scatter diagonally: score col j = c - 15 + row_in_tile.
  if (qt >= 4 && qt < 68) {
    int i0 = (qt - 4) * 16;
    int w0 = 1008 - i0;
    v16h c0 = load_A(qv, 64, i0, 0, lane);
    v16h c1 = load_A(qv, 64, i0, 32, lane);
#if USE_ASYNC_LDS
    {
      int nt = 0;
      stage_tile(ph + (size_t)(w0 + wave * 16) * 64, 0);
      for (int t = wave; t < 65; t += 4) {
        if (t + 4 < 65) {
          stage_tile(ph + (size_t)(w0 + (t + 4) * 16) * 64, nt ^ 1);
          __builtin_amdgcn_s_wait_asynccnt(4);
        } else {
          __builtin_amdgcn_s_wait_asynccnt(0);
        }
        const _Float16* bsrc = &stage[wave][nt][0];
        v8f acc = {};
        acc = wmma_f16f32(c0, load_B_T(bsrc, 64, 0, 0, lane), acc);
        acc = wmma_f16f32(c1, load_B_T(bsrc, 64, 0, 32, lane), acc);
        int c = t * 16 + (lane & 15);          // window column
#pragma unroll
        for (int j = 0; j < 8; ++j) {
          int ih = j + ((lane >> 4) << 3);     // row within tile
          int jj = c - 15 + ih;                // utterance kv column
          if (jj >= 0 && jj < 1024)
            sS[ih * SLD + (15 + 64 + jj)] += acc[j];   // M+R = 79 offset
        }
        nt ^= 1;
      }
    }
#else
    for (int t = wave; t < 65; t += 4) {
      int p0 = w0 + t * 16;
      v8f acc = {};
      acc = wmma_f16f32(c0, load_B_T(ph, 64, p0, 0, lane), acc);
      acc = wmma_f16f32(c1, load_B_T(ph, 64, p0, 32, lane), acc);
      int c = t * 16 + (lane & 15);
#pragma unroll
      for (int j = 0; j < 8; ++j) {
        int ih = j + ((lane >> 4) << 3);
        int jj = c - 15 + ih;
        if (jj >= 0 && jj < 1024)
          sS[ih * SLD + (15 + 64 + jj)] += acc[j];
      }
    }
#endif
  }
  __syncthreads();

  // ---- Phase 3: analytic mask + scale + softmax; probs -> f16 LDS ---------
  // 8 consecutive lanes own one row (rows 0..15 over 128 threads).
  {
    int row = tid >> 3, sub = tid & 7;
    int q = q0 + row;
    int chunk = q < 64 ? (q >> 2) : (q < 1088 ? ((q - 64) >> 6) : (q - 1088));
    int padstart = 79 + lengths[b];          // kv >= padstart -> masked

    float mx = -3.4e38f;
    for (int kv = sub; kv < SLD; kv += 8) {
      float v;
      if (kv >= KVLEN) {
        v = -__builtin_inff();               // artificial pad columns
      } else {
        bool allowed;
        if (kv < 15)      allowed = (kv < chunk);
        else if (kv < 79) allowed = (((kv - 15) >> 2) == chunk);
        else {
          int uu = kv - 79;
          int lo = (chunk == 0) ? 0 : (chunk - 1) * 64;
          allowed = (uu >= lo) && (uu < (chunk + 1) * 64);
        }
        v = (!allowed || kv >= padstart) ? NEG_INF_VAL
                                         : sS[row * SLD + kv] * 0.125f;
      }
      sS[row * SLD + kv] = v;
      mx = fmaxf(mx, v);
    }
#pragma unroll
    for (int off = 4; off; off >>= 1) mx = fmaxf(mx, __shfl_xor(mx, off, 8));

    float sum = 0.0f;
    for (int kv = sub; kv < SLD; kv += 8) {
      float e = __expf(sS[row * SLD + kv] - mx);   // exp(-inf)=0 on pad cols
      sum += e;
      sP[row * SLD + kv] = (_Float16)e;
    }
#pragma unroll
    for (int off = 4; off; off >>= 1) sum += __shfl_xor(sum, off, 8);
    float inv = 1.0f / sum;
    for (int kv = sub; kv < SLD; kv += 8)
      sP[row * SLD + kv] = (_Float16)((float)sP[row * SLD + kv] * inv);
  }
  __syncthreads();

  // ---- Phase 4: attn tile = probs[16][1120] @ V[1120][64]; wave owns 16 cols
  // V stored transposed per head (VT[hd][kv]) so B-fragments are contiguous.
  {
    int n0 = wave * 16;
    v8f acc = {};
    for (int k0 = 0; k0 < KVPAD; k0 += 32)
      acc = wmma_f16f32(load_A(sP, SLD, 0, k0, lane),
                        load_B_T(vt, KVPAD, n0, k0, lane), acc);
    int col = n0 + (lane & 15);
    int rb = (lane >> 4) << 3;
#pragma unroll
    for (int j = 0; j < 8; ++j) {
      int qq = q0 + rb + j;
      attn_out[((size_t)qq * 8 + b) * 512 + h * 64 + col] = acc[j];
    }
  }
}

// ---------------------------------------------------------------------------
// Elementwise helpers
// ---------------------------------------------------------------------------
__global__ __launch_bounds__(256) void cvt_f32_f16(const float* __restrict__ s,
                                                   _Float16* __restrict__ d, int n) {
  int i = blockIdx.x * 256 + threadIdx.x;
  if (i < n) d[i] = (_Float16)s[i];
}

__global__ __launch_bounds__(256) void cvt_f32_f16_pad(const float* __restrict__ s,
                                                       _Float16* __restrict__ d,
                                                       int n, int ntot) {
  int i = blockIdx.x * 256 + threadIdx.x;
  if (i < ntot) d[i] = (i < n) ? (_Float16)s[i] : (_Float16)0.0f;
}

// Xq rows (t*8+b): concat(right_context[64], utterance[1024], summary[16])
__global__ __launch_bounds__(256) void build_xq(const float* __restrict__ rc,
                                                const float* __restrict__ utt,
                                                const float* __restrict__ sm,
                                                _Float16* __restrict__ dst) {
  int i = blockIdx.x * 256 + threadIdx.x;
  if (i >= 8832 * 512) return;
  int d = i & 511, row = i >> 9, t = row >> 3, b = row & 7;
  float v;
  if (t < 64)        v = rc [((size_t)t * 8 + b) * 512 + d];
  else if (t < 1088) v = utt[((size_t)(t - 64) * 8 + b) * 512 + d];
  else               v = sm [((size_t)(t - 1088) * 8 + b) * 512 + d];
  dst[i] = (_Float16)v;
}

// Xkv rows: concat(memory[15], right_context[64], utterance[1024]) + zero pad
__global__ __launch_bounds__(256) void build_xkv(const float* __restrict__ mem,
                                                 const float* __restrict__ rc,
                                                 const float* __restrict__ utt,
                                                 _Float16* __restrict__ dst) {
  int i = blockIdx.x * 256 + threadIdx.x;
  if (i >= 8832 * 512) return;
  int d = i & 511, row = i >> 9, t = row >> 3, b = row & 7;
  float v = 0.0f;
  if (t < 15)        v = mem[((size_t)t * 8 + b) * 512 + d];
  else if (t < 79)   v = rc [((size_t)(t - 15) * 8 + b) * 512 + d];
  else if (t < 1103) v = utt[((size_t)(t - 79) * 8 + b) * 512 + d];
  dst[i] = (_Float16)v;
}

// QU[bh][q][e] = q_proj + pos_bias_u ; QV[bh][i][e] = q_proj(utt) + pos_bias_v
__global__ __launch_bounds__(256) void build_q_heads(const float* __restrict__ qf,
                                                     const float* __restrict__ pbu,
                                                     const float* __restrict__ pbv,
                                                     _Float16* __restrict__ QU,
                                                     _Float16* __restrict__ QV) {
  int i = blockIdx.x * 256 + threadIdx.x;
  if (i >= 64 * QLEN * 64) return;
  int e = i & 63, q = (i >> 6) % QLEN, bh = i / (QLEN * 64);
  int b = bh >> 3, hh = bh & 7;
  float qval = qf[((size_t)q * 8 + b) * 512 + hh * 64 + e];
  QU[i] = (_Float16)(qval + pbu[hh * 64 + e]);
  if (q >= 64 && q < 1088)
    QV[((size_t)bh * 1024 + (q - 64)) * 64 + e] = (_Float16)(qval + pbv[hh * 64 + e]);
}

// Kh[bh][kv][e] and VT[bh][e][kv] (transposed V) from kv proj [kv][b][1024]
__global__ __launch_bounds__(256) void build_kv_heads(const float* __restrict__ kvf,
                                                      _Float16* __restrict__ Kh,
                                                      _Float16* __restrict__ VT) {
  int i = blockIdx.x * 256 + threadIdx.x;
  if (i >= 64 * KVPAD * 64) return;
  int e = i & 63, kv = (i >> 6) % KVPAD, bh = i / (KVPAD * 64);
  int b = bh >> 3, hh = bh & 7;
  float kval = 0.0f, vval = 0.0f;
  if (kv < KVLEN) {
    size_t base = ((size_t)kv * 8 + b) * 1024 + hh * 64 + e;
    kval = kvf[base];
    vval = kvf[base + 512];
  }
  if (kv < QLEN) Kh[((size_t)bh * QLEN + kv) * 64 + e] = (_Float16)kval;
  VT[((size_t)bh * 64 + e) * KVPAD + kv] = (_Float16)vval;
}

// Ph[h][p][e] from pos projection [p][512] (row 2047 already zero)
__global__ __launch_bounds__(256) void build_p_heads(const float* __restrict__ pf,
                                                     _Float16* __restrict__ Ph) {
  int i = blockIdx.x * 256 + threadIdx.x;
  if (i >= 8 * PEPAD * 64) return;
  int e = i & 63, p = (i >> 6) % PEPAD, hh = i / (PEPAD * 64);
  Ph[i] = (_Float16)pf[(size_t)p * 512 + hh * 64 + e];
}

__global__ __launch_bounds__(256) void clip_copy(const float* __restrict__ s,
                                                 float* __restrict__ d, int n) {
  int i = blockIdx.x * 256 + threadIdx.x;
  if (i < n) d[i] = fminf(10.0f, fmaxf(-10.0f, s[i]));
}

// ---------------------------------------------------------------------------
extern "C" void kernel_launch(void* const* d_in, const int* in_sizes, int n_in,
                              void* d_out, int out_size, void* d_ws, size_t ws_size,
                              hipStream_t stream) {
  (void)in_sizes; (void)n_in; (void)out_size; (void)ws_size;
  const float* utterance = (const float*)d_in[0];
  const int*   lengths   = (const int*)  d_in[1];
  const float* rcx       = (const float*)d_in[2];
  const float* summary   = (const float*)d_in[3];
  const float* memry     = (const float*)d_in[4];
  /* d_in[5] attention_mask: recomputed analytically in-kernel */
  const float* pos_emb   = (const float*)d_in[6];
  const float* W_kv      = (const float*)d_in[7];
  const float* b_kv      = (const float*)d_in[8];
  const float* W_q       = (const float*)d_in[9];
  const float* b_q       = (const float*)d_in[10];
  const float* W_out     = (const float*)d_in[11];
  const float* b_out     = (const float*)d_in[12];
  const float* W_pos     = (const float*)d_in[13];
  const float* pbu       = (const float*)d_in[14];
  const float* pbv       = (const float*)d_in[15];
  float* out = (float*)d_out;

  char* ws = (char*)d_ws;
  size_t off = 0;
  auto alloc = [&](size_t bytes) -> char* {
    char* p = ws + off;
    off += (bytes + 255) & ~(size_t)255;
    return p;
  };
  _Float16* Xq_h   = (_Float16*)alloc((size_t)8832 * 512 * 2);
  _Float16* Xkv_h  = (_Float16*)alloc((size_t)8832 * 512 * 2);
  _Float16* Wq_h   = (_Float16*)alloc((size_t)512 * 512 * 2);
  _Float16* Wkv_h  = (_Float16*)alloc((size_t)1024 * 512 * 2);
  _Float16* Wpos_h = (_Float16*)alloc((size_t)512 * 512 * 2);
  _Float16* Wout_h = (_Float16*)alloc((size_t)512 * 512 * 2);
  _Float16* Pin_h  = (_Float16*)alloc((size_t)PEPAD * 512 * 2);
  float*    q_f    = (float*)   alloc((size_t)8832 * 512 * 4);
  float*    kv_f   = (float*)   alloc((size_t)8832 * 1024 * 4);
  float*    pos_f  = (float*)   alloc((size_t)PEPAD * 512 * 4);
  _Float16* QU     = (_Float16*)alloc((size_t)64 * QLEN * 64 * 2);
  _Float16* QV     = (_Float16*)alloc((size_t)64 * 1024 * 64 * 2);
  _Float16* Kh     = (_Float16*)alloc((size_t)64 * QLEN * 64 * 2);
  _Float16* VT     = (_Float16*)alloc((size_t)64 * 64 * KVPAD * 2);
  _Float16* Ph     = (_Float16*)alloc((size_t)8 * PEPAD * 64 * 2);
  float*    attn_f = (float*)   alloc((size_t)QLEN * 8 * 512 * 4);
  _Float16* attn_h = (_Float16*)alloc((size_t)8832 * 512 * 2);

  auto blk = [](int n) { return (n + 255) / 256; };

  // f16 operand prep
  build_xq <<<blk(8832 * 512), 256, 0, stream>>>(rcx, utterance, summary, Xq_h);
  build_xkv<<<blk(8832 * 512), 256, 0, stream>>>(memry, rcx, utterance, Xkv_h);
  cvt_f32_f16<<<blk(512 * 512),  256, 0, stream>>>(W_q,   Wq_h,   512 * 512);
  cvt_f32_f16<<<blk(1024 * 512), 256, 0, stream>>>(W_kv,  Wkv_h,  1024 * 512);
  cvt_f32_f16<<<blk(512 * 512),  256, 0, stream>>>(W_pos, Wpos_h, 512 * 512);
  cvt_f32_f16<<<blk(512 * 512),  256, 0, stream>>>(W_out, Wout_h, 512 * 512);
  cvt_f32_f16_pad<<<blk(PEPAD * 512), 256, 0, stream>>>(pos_emb, Pin_h,
                                                        2047 * 512, PEPAD * 512);

  // Projections (WMMA GEMMs, 16x64 per wave)
  gemm_xwT<<<dim3(552, 8),  32, 0, stream>>>(Xq_h,  Wq_h,  b_q,     q_f,  512,  512);
  gemm_xwT<<<dim3(552, 16), 32, 0, stream>>>(Xkv_h, Wkv_h, b_kv,    kv_f, 1024, 512);
  gemm_xwT<<<dim3(128, 8),  32, 0, stream>>>(Pin_h, Wpos_h, nullptr, pos_f, 512, 512);

  // Head split + bias folding (f16 attention operands)
  build_q_heads <<<blk(64 * QLEN * 64),  256, 0, stream>>>(q_f, pbu, pbv, QU, QV);
  build_kv_heads<<<blk(64 * KVPAD * 64), 256, 0, stream>>>(kv_f, Kh, VT);
  build_p_heads <<<blk(8 * PEPAD * 64),  256, 0, stream>>>(pos_f, Ph);

  // Fused attention: 69 q-tiles x 64 (b,h); ~105 KB dynamic LDS + 16 KB stage
  size_t lds = (size_t)16 * SLD * 4 + (size_t)16 * SLD * 2;
  attention_kernel<<<dim3(69, 64), 128, lds, stream>>>(QU, QV, Kh, VT, Ph,
                                                       lengths, attn_f);

  // Output projection for rows [0, R+U) and clipped memory output
  cvt_f32_f16<<<blk(8832 * 512), 256, 0, stream>>>(attn_f, attn_h, 8832 * 512);
  gemm_xwT<<<dim3(544, 8), 32, 0, stream>>>(attn_h, Wout_h, b_out, out, 512, 512);
  clip_copy<<<blk(16 * 8 * 512), 256, 0, stream>>>(attn_f + (size_t)8704 * 512,
                                                   out + (size_t)8704 * 512,
                                                   16 * 8 * 512);
}
